// LinearAttention_996432413010
// MI455X (gfx1250) — matrix operand order repaired and checked by reading, not compiled, not run on hardware
//
#include <hip/hip_runtime.h>
#include <stdint.h>

// ---------------- problem constants ----------------
constexpr int BATCH  = 4;
constexpr int SEQ    = 4096;
constexpr int HIDDIM = 2048;
constexpr int NHEADS = 16;
constexpr int HD     = 128;   // head dim
constexpr int CK     = 64;    // FLA chunk

typedef __bf16 bf16_t;
typedef __attribute__((ext_vector_type(16))) __bf16 v16bf;
typedef __attribute__((ext_vector_type(8)))  float  v8f;
typedef __attribute__((ext_vector_type(4)))  int    v4i;

// address-space-qualified pointers for the CDNA5 async-to-LDS builtin
typedef v4i __attribute__((address_space(1)))* gv4i_ptr;   // global
typedef v4i __attribute__((address_space(3)))* sv4i_ptr;   // LDS

#define WMMA_BF16(a,b,c) \
  __builtin_amdgcn_wmma_f32_16x16x32_bf16(false,(a),false,(b),(short)0,(c),false,false)

// CDNA5 async global->LDS (ASYNCcnt-tracked) if this toolchain exposes it.
#if defined(__has_builtin)
#if __has_builtin(__builtin_amdgcn_global_load_async_to_lds_b128) && \
    __has_builtin(__builtin_amdgcn_s_wait_asynccnt)
#define HAVE_ASYNC_LDS 1
#endif
#endif
#ifndef HAVE_ASYNC_LDS
#define HAVE_ASYNC_LDS 0
#endif

static __device__ __forceinline__ bf16_t f2bf(float f) {
  uint32_t u = __builtin_bit_cast(uint32_t, f);
  uint32_t r = u + 0x7FFFu + ((u >> 16) & 1u);          // round-to-nearest-even
  return __builtin_bit_cast(bf16_t, (uint16_t)(r >> 16));
}

// A fragment (16x32 bf16, M x K). Per ISA 7.12.2: lane m=L&15, half=L>>4;
// VGPR0-3 hold K = half*8 + [0,8), VGPR4-7 hold K = 16 + half*8 + [0,8).
static __device__ __forceinline__ v16bf load_frag_a(const bf16_t* base, int ld) {
  int lane = threadIdx.x & 31;
  int m = lane & 15, half = lane >> 4;
  const bf16_t* row = base + m * ld;
  union { v16bf v; uint4 u[2]; } f;
  f.u[0] = *(const uint4*)(row + half * 8);
  f.u[1] = *(const uint4*)(row + 16 + half * 8);
  return f.v;
}

// B fragment (32x16 bf16, K x N), stored N-major ([n][k]). Lane n=L&15 holds
// K = half*8 + [0,8) then K = 16 + half*8 + [0,8) (same order as A).
static __device__ __forceinline__ v16bf load_frag_b(const bf16_t* base, int ld) {
  int lane = threadIdx.x & 31;
  int n = lane & 15, half = lane >> 4;
  const bf16_t* row = base + n * ld;
  union { v16bf v; uint4 u[2]; } f;
  f.u[0] = *(const uint4*)(row + half * 8);
  f.u[1] = *(const uint4*)(row + 16 + half * 8);
  return f.v;
}

// ---------------- elementwise f32 -> bf16 ----------------
__global__ __launch_bounds__(256) void k_f32_to_bf16(const float* __restrict__ in,
                                                     bf16_t* __restrict__ out,
                                                     long long n) {
  long long i = ((long long)blockIdx.x * blockDim.x + threadIdx.x) * 4;
  if (i + 3 < n) {
    float4 v = *(const float4*)(in + i);
    union { uint2 u; bf16_t e[4]; } p;
    p.e[0] = f2bf(v.x); p.e[1] = f2bf(v.y); p.e[2] = f2bf(v.z); p.e[3] = f2bf(v.w);
    *(uint2*)(out + i) = p.u;
  }
}

// ---------------- f32 [R][C] -> bf16 [C][R] (for GEMM B-operands) ----------------
__global__ __launch_bounds__(256) void k_transpose_bf16(const float* __restrict__ in,
                                                        bf16_t* __restrict__ out,
                                                        int R, int C) {
  __shared__ bf16_t tile[64][65];
  int br = blockIdx.y * 64, bc = blockIdx.x * 64;
  int tid = threadIdx.x;
  int r = tid >> 2, c0 = (tid & 3) * 16;
  for (int i = 0; i < 16; ++i)
    tile[r][c0 + i] = f2bf(in[(long long)(br + r) * C + bc + c0 + i]);
  __syncthreads();
  int co = tid >> 2, r0 = (tid & 3) * 16;
  for (int i = 0; i < 16; ++i)
    out[(long long)(bc + co) * R + br + r0 + i] = tile[r0 + i][co];
}

// ---------------- bf16 WMMA GEMM: C[M,N] = A[M,K] @ Bt[N,K]^T ----------------
// grid = (N/128, M/128), block = 256 (8 waves, 2x4), 128x128 output tile.
// Double-buffered LDS, BK=64: one barrier per K-step; staging for step s+1
// overlaps the 16 WMMAs of step s (async-to-LDS when available).
__global__ __launch_bounds__(256) void k_gemm_bf16(const bf16_t* __restrict__ A,
                                                   const bf16_t* __restrict__ Bt,
                                                   float* __restrict__ Cf,
                                                   bf16_t* __restrict__ Cb,
                                                   int M, int N, int K) {
  constexpr int BK = 64, LD = 72;               // 144B rows: aligned, conflict-free
  __shared__ bf16_t As[2][128 * LD];
  __shared__ bf16_t Bs[2][128 * LD];
  int tid = threadIdx.x, w = tid >> 5, lane = tid & 31;
  int wm = w >> 2, wn = w & 3;                  // wave grid 2 (M) x 4 (N)
  long long bm = (long long)blockIdx.y * 128, bn = (long long)blockIdx.x * 128;

  v8f acc[4][2] = {};
  int lr = tid >> 1, seg = (tid & 1) * 32;      // thread stages 32 elems per tile
  const bf16_t* ga = A  + (bm + lr) * K + seg;
  const bf16_t* gb = Bt + (bn + lr) * K + seg;
  const int soff = lr * LD + seg;
  const int nsteps = K / BK;

#if HAVE_ASYNC_LDS
  auto issue = [&](int buf, int k0) {
    const bf16_t* gA = ga + k0;
    const bf16_t* gB = gb + k0;
    bf16_t* lA = &As[buf][soff];
    bf16_t* lB = &Bs[buf][soff];
    for (int i = 0; i < 4; ++i) {
      __builtin_amdgcn_global_load_async_to_lds_b128(
          (gv4i_ptr)(gA + i * 8), (sv4i_ptr)(lA + i * 8), 0, 0);
      __builtin_amdgcn_global_load_async_to_lds_b128(
          (gv4i_ptr)(gB + i * 8), (sv4i_ptr)(lB + i * 8), 0, 0);
    }
  };
  issue(0, 0);
  __builtin_amdgcn_s_wait_asynccnt(0);
  __syncthreads();
  for (int s = 0; s < nsteps; ++s) {
    int cur = s & 1;
    if (s + 1 < nsteps) {
      issue(cur ^ 1, (s + 1) * BK);
      if (s + 2 < nsteps) {
        __builtin_prefetch(ga + (s + 2) * BK, 0, 1);
        __builtin_prefetch(gb + (s + 2) * BK, 0, 1);
      }
    }
    for (int kk = 0; kk < BK; kk += 32) {
      v16bf af[4];
      for (int i = 0; i < 4; ++i)
        af[i] = load_frag_a(&As[cur][(wm * 64 + i * 16) * LD + kk], LD);
      for (int j = 0; j < 2; ++j) {
        v16bf bfr = load_frag_b(&Bs[cur][(wn * 32 + j * 16) * LD + kk], LD);
        for (int i = 0; i < 4; ++i)
          acc[i][j] = WMMA_BF16(af[i], bfr, acc[i][j]);
      }
    }
    if (s + 1 < nsteps) __builtin_amdgcn_s_wait_asynccnt(0);
    __syncthreads();
  }
#else
  uint4 ra[4], rb[4];
  auto gload = [&](int k0) {
    for (int i = 0; i < 4; ++i) {
      ra[i] = *(const uint4*)(ga + k0 + i * 8);
      rb[i] = *(const uint4*)(gb + k0 + i * 8);
    }
  };
  auto sstore = [&](int buf) {
    uint4* lA = (uint4*)&As[buf][soff];
    uint4* lB = (uint4*)&Bs[buf][soff];
    for (int i = 0; i < 4; ++i) { lA[i] = ra[i]; lB[i] = rb[i]; }
  };
  gload(0);
  sstore(0);
  __syncthreads();
  for (int s = 0; s < nsteps; ++s) {
    int cur = s & 1;
    if (s + 1 < nsteps) {
      gload((s + 1) * BK);
      if (s + 2 < nsteps) {
        __builtin_prefetch(ga + (s + 2) * BK, 0, 1);
        __builtin_prefetch(gb + (s + 2) * BK, 0, 1);
      }
    }
    for (int kk = 0; kk < BK; kk += 32) {
      v16bf af[4];
      for (int i = 0; i < 4; ++i)
        af[i] = load_frag_a(&As[cur][(wm * 64 + i * 16) * LD + kk], LD);
      for (int j = 0; j < 2; ++j) {
        v16bf bfr = load_frag_b(&Bs[cur][(wn * 32 + j * 16) * LD + kk], LD);
        for (int i = 0; i < 4; ++i)
          acc[i][j] = WMMA_BF16(af[i], bfr, acc[i][j]);
      }
    }
    if (s + 1 < nsteps) sstore(cur ^ 1);
    __syncthreads();
  }
#endif

  for (int i = 0; i < 4; ++i)
    for (int j = 0; j < 2; ++j) {
      long long row0 = bm + wm * 64 + i * 16 + ((lane >> 4) << 3);
      long long col  = bn + wn * 32 + j * 16 + (lane & 15);
      if (Cf) { for (int g = 0; g < 8; ++g) Cf[(row0 + g) * N + col] = acc[i][j][g]; }
      else    { for (int g = 0; g < 8; ++g) Cb[(row0 + g) * N + col] = f2bf(acc[i][j][g]); }
    }
}

// ---------------- fused feature map: Y = (X@W1 + b1) * (X@W2 + b2) * scale ----------------
// X: [M, 128] bf16 (M = B*T*NH rows), W*t: [128][128] bf16 (transposed). grid = M/64.
__global__ __launch_bounds__(256) void k_featmap(const bf16_t* __restrict__ X,
                                                 const bf16_t* __restrict__ W1t,
                                                 const bf16_t* __restrict__ W2t,
                                                 const float* __restrict__ b1,
                                                 const float* __restrict__ b2,
                                                 bf16_t* __restrict__ Y,
                                                 float scale) {
  constexpr int LD = 136;                       // 272B rows, aligned + conflict-free
  __shared__ bf16_t Xs [64  * LD];
  __shared__ bf16_t W1s[128 * LD];
  __shared__ bf16_t W2s[128 * LD];
  int tid = threadIdx.x, w = tid >> 5, lane = tid & 31;
  long long bm = (long long)blockIdx.x * 64;
  {
    int r = tid >> 2, c0 = (tid & 3) * 32;
    const uint4* gx = (const uint4*)(X + (bm + r) * HD + c0);
    uint4* sx = (uint4*)(Xs + r * LD + c0);
    for (int i = 0; i < 4; ++i) sx[i] = gx[i];
    int wr = tid >> 1, wc = (tid & 1) * 64;
    const uint4* g1 = (const uint4*)(W1t + (long long)wr * HD + wc);
    const uint4* g2 = (const uint4*)(W2t + (long long)wr * HD + wc);
    uint4* s1 = (uint4*)(W1s + wr * LD + wc);
    uint4* s2 = (uint4*)(W2s + wr * LD + wc);
    for (int i = 0; i < 8; ++i) { s1[i] = g1[i]; s2[i] = g2[i]; }
  }
  __syncthreads();

  int wmt = w & 3, wn0 = (w >> 2) * 64;         // wave: 1 M-tile x 4 N-tiles, two mats
  v8f a1[4] = {}, a2[4] = {};
  for (int ks = 0; ks < 128; ks += 32) {
    v16bf af = load_frag_a(Xs + wmt * 16 * LD + ks, LD);
    for (int j = 0; j < 4; ++j) {
      a1[j] = WMMA_BF16(af, load_frag_b(W1s + (wn0 + j * 16) * LD + ks, LD), a1[j]);
      a2[j] = WMMA_BF16(af, load_frag_b(W2s + (wn0 + j * 16) * LD + ks, LD), a2[j]);
    }
  }
  for (int j = 0; j < 4; ++j) {
    int col = wn0 + j * 16 + (lane & 15);
    long long row0 = bm + wmt * 16 + ((lane >> 4) << 3);
    float bb1 = b1[col], bb2 = b2[col];
    for (int g = 0; g < 8; ++g) {
      float v = (a1[j][g] + bb1) * (a2[j][g] + bb2) * scale;
      Y[(row0 + g) * HD + col] = f2bf(v);
    }
  }
}

// ---------------- chunked causal linear attention ----------------
// One workgroup per (b,h). 128x128 fp32 KV state lives in WMMA accumulators
// (8 tiles per wave x 8 waves); bf16 S^T is re-materialized to LDS each chunk
// for the q@S inter-chunk term. q/k/v layout: [B,T,NH,HD] bf16.
__global__ __launch_bounds__(256) void k_attn(const bf16_t* __restrict__ qf,
                                              const bf16_t* __restrict__ kf,
                                              const bf16_t* __restrict__ vf,
                                              float* __restrict__ o) {
  constexpr int LDQ = 136, LDT = 72;
  __shared__ bf16_t q_s [CK * LDQ];             // [t][d]
  __shared__ bf16_t k_s [CK * LDQ];             // [t][d]  (B-op for q k^T)
  __shared__ bf16_t kT_s[HD * LDT];             // [d][t]  (A-op for k^T v)
  __shared__ bf16_t vT_s[HD * LDT];             // [d][t]  (B-op everywhere v appears)
  __shared__ bf16_t att_s[CK * LDT];            // [t][s]
  __shared__ bf16_t S_s [HD * LDQ];             // S^T: [dv][dk] (B-op for q@S)

  int tid = threadIdx.x, w = tid >> 5, lane = tid & 31;
  int b = blockIdx.x / NHEADS, h = blockIdx.x % NHEADS;

  for (int i = tid; i < HD * LDQ / 2; i += 256) ((uint32_t*)S_s)[i] = 0u;
  v8f Sacc[8] = {};                             // state tiles: rows w*16.., all dv

  int r = tid >> 2, seg = (tid & 3) * 32;
  for (int c = 0; c < SEQ / CK; ++c) {
    __syncthreads();
    { // stage q,k,v chunk; build transposed copies for A/B operand layouts
      long long grow = ((long long)b * SEQ + (long long)c * CK + r) * NHEADS + h;
      const uint4* gq = (const uint4*)(qf + grow * HD + seg);
      const uint4* gk = (const uint4*)(kf + grow * HD + seg);
      const uint4* gv = (const uint4*)(vf + grow * HD + seg);
      uint4* sq = (uint4*)(q_s + r * LDQ + seg);
      uint4* sk = (uint4*)(k_s + r * LDQ + seg);
      union { uint4 u[4]; bf16_t e[32]; } kb, vb;
      for (int i = 0; i < 4; ++i) { sq[i] = gq[i]; kb.u[i] = gk[i]; vb.u[i] = gv[i]; sk[i] = kb.u[i]; }
      for (int i = 0; i < 32; ++i) {
        kT_s[(seg + i) * LDT + r] = kb.e[i];
        vT_s[(seg + i) * LDT + r] = vb.e[i];
      }
    }
    __syncthreads();

    // phase 1: att = q k^T (64x64), causal mask, write bf16 A-operand to LDS
    for (int p = 0; p < 2; ++p) {
      int idx = w * 2 + p, ti = idx >> 2, tj = idx & 3;
      v8f a = {};
      for (int ks = 0; ks < 128; ks += 32)
        a = WMMA_BF16(load_frag_a(q_s + ti * 16 * LDQ + ks, LDQ),
                      load_frag_b(k_s + tj * 16 * LDQ + ks, LDQ), a);
      int n  = tj * 16 + (lane & 15);
      int m0 = ti * 16 + ((lane >> 4) << 3);
      for (int g = 0; g < 8; ++g) {
        int m = m0 + g;
        att_s[m * LDT + n] = f2bf(n <= m ? a[g] : 0.f);
      }
    }
    __syncthreads();

    // phase 2: o_c = att@v (intra) + q@S (inter); store fp32
    {
      int ti = w >> 1;
      for (int jj = 0; jj < 4; ++jj) {
        int tj = (w & 1) * 4 + jj;
        v8f a = {};
        for (int ks = 0; ks < 64; ks += 32)
          a = WMMA_BF16(load_frag_a(att_s + ti * 16 * LDT + ks, LDT),
                        load_frag_b(vT_s + tj * 16 * LDT + ks, LDT), a);
        for (int ks = 0; ks < 128; ks += 32)
          a = WMMA_BF16(load_frag_a(q_s + ti * 16 * LDQ + ks, LDQ),
                        load_frag_b(S_s + tj * 16 * LDQ + ks, LDQ), a);
        int n  = tj * 16 + (lane & 15);
        int m0 = ti * 16 + ((lane >> 4) << 3);
        for (int g = 0; g < 8; ++g) {
          long long t = (long long)c * CK + m0 + g;
          o[(((long long)b * SEQ + t) * NHEADS + h) * HD + n] = a[g];
        }
      }
    }
    __syncthreads();

    // phase 3: S += k^T v (state stays fp32 in accumulators); emit bf16 S^T
    {
      v16bf ka0 = load_frag_a(kT_s + w * 16 * LDT + 0,  LDT);
      v16bf ka1 = load_frag_a(kT_s + w * 16 * LDT + 32, LDT);
      for (int j = 0; j < 8; ++j) {
        Sacc[j] = WMMA_BF16(ka0, load_frag_b(vT_s + j * 16 * LDT + 0,  LDT), Sacc[j]);
        Sacc[j] = WMMA_BF16(ka1, load_frag_b(vT_s + j * 16 * LDT + 32, LDT), Sacc[j]);
      }
      for (int j = 0; j < 8; ++j) {             // transposed write => packed 16B stores
        int dv  = j * 16 + (lane & 15);
        int dk0 = w * 16 + ((lane >> 4) << 3);
        union { uint4 u; bf16_t e[8]; } p;
        for (int g = 0; g < 8; ++g) p.e[g] = f2bf(Sacc[j][g]);
        *(uint4*)(S_s + dv * LDQ + dk0) = p.u;
      }
    }
  }
}

// ---------------- RMSNorm over HD (one wave32 per row) ----------------
__global__ __launch_bounds__(256) void k_rmsnorm(const float* __restrict__ o,
                                                 const float* __restrict__ g,
                                                 bf16_t* __restrict__ on) {
  long long row = (long long)blockIdx.x * 8 + (threadIdx.x >> 5);
  int lane = threadIdx.x & 31;
  float4 v = *(const float4*)(o + row * HD + lane * 4);
  float ss = v.x * v.x + v.y * v.y + v.z * v.z + v.w * v.w;
  for (int m = 16; m >= 1; m >>= 1) ss += __shfl_xor(ss, m, 32);
  float rr = rsqrtf(ss * (1.f / HD) + 1e-5f);
  float4 gg = *(const float4*)(g + lane * 4);
  union { uint2 u; bf16_t e[4]; } p;
  p.e[0] = f2bf(v.x * rr * gg.x); p.e[1] = f2bf(v.y * rr * gg.y);
  p.e[2] = f2bf(v.z * rr * gg.z); p.e[3] = f2bf(v.w * rr * gg.w);
  *(uint2*)(on + row * HD + lane * 4) = p.u;
}

// ---------------- host orchestration ----------------
extern "C" void kernel_launch(void* const* d_in, const int* in_sizes, int n_in,
                              void* d_out, int out_size, void* d_ws, size_t ws_size,
                              hipStream_t stream) {
  (void)in_sizes; (void)n_in; (void)out_size; (void)ws_size;
  const float* x      = (const float*)d_in[0];
  const float* wq     = (const float*)d_in[1];
  const float* wk     = (const float*)d_in[2];
  const float* wv     = (const float*)d_in[3];
  const float* fmq1_w = (const float*)d_in[4];  const float* fmq1_b = (const float*)d_in[5];
  const float* fmq2_w = (const float*)d_in[6];  const float* fmq2_b = (const float*)d_in[7];
  const float* fmk1_w = (const float*)d_in[8];  const float* fmk1_b = (const float*)d_in[9];
  const float* fmk2_w = (const float*)d_in[10]; const float* fmk2_b = (const float*)d_in[11];
  const float* gvec   = (const float*)d_in[12];
  const float* wo     = (const float*)d_in[13];
  float* out = (float*)d_out;

  char* ws = (char*)d_ws;
  size_t off = 0;
  auto alloc = [&](size_t bytes) -> char* {
    char* p = ws + off; off += (bytes + 255) & ~(size_t)255; return p;
  };
  const size_t MT = (size_t)BATCH * SEQ;                 // 16384 token rows
  bf16_t* xb   = (bf16_t*)alloc(MT * HIDDIM * 2);
  bf16_t* wqT  = (bf16_t*)alloc((size_t)HIDDIM * HIDDIM * 2);
  bf16_t* wkT  = (bf16_t*)alloc((size_t)HIDDIM * HIDDIM * 2);
  bf16_t* wvT  = (bf16_t*)alloc((size_t)HIDDIM * HIDDIM * 2);
  bf16_t* woT  = (bf16_t*)alloc((size_t)HIDDIM * HIDDIM * 2);
  bf16_t* fq1T = (bf16_t*)alloc((size_t)HD * HD * 2);
  bf16_t* fq2T = (bf16_t*)alloc((size_t)HD * HD * 2);
  bf16_t* fk1T = (bf16_t*)alloc((size_t)HD * HD * 2);
  bf16_t* fk2T = (bf16_t*)alloc((size_t)HD * HD * 2);
  bf16_t* qproj = (bf16_t*)alloc(MT * HIDDIM * 2);
  bf16_t* kproj = (bf16_t*)alloc(MT * HIDDIM * 2);
  bf16_t* vproj = (bf16_t*)alloc(MT * HIDDIM * 2);
  bf16_t* qfeat = (bf16_t*)alloc(MT * HIDDIM * 2);
  bf16_t* kfeat = (bf16_t*)alloc(MT * HIDDIM * 2);
  float*  ofp   = (float*)qproj;    // reuse: qproj+kproj (128MB) dead after featmap
  bf16_t* onorm = kfeat;            // reuse: kfeat dead after attention

  long long nx = (long long)MT * HIDDIM;
  k_f32_to_bf16<<<(uint32_t)(nx / 1024), 256, 0, stream>>>(x, xb, nx);

  dim3 tg(HIDDIM / 64, HIDDIM / 64);
  k_transpose_bf16<<<tg, 256, 0, stream>>>(wq, wqT, HIDDIM, HIDDIM);
  k_transpose_bf16<<<tg, 256, 0, stream>>>(wk, wkT, HIDDIM, HIDDIM);
  k_transpose_bf16<<<tg, 256, 0, stream>>>(wv, wvT, HIDDIM, HIDDIM);
  k_transpose_bf16<<<tg, 256, 0, stream>>>(wo, woT, HIDDIM, HIDDIM);
  dim3 tf(HD / 64, HD / 64);
  k_transpose_bf16<<<tf, 256, 0, stream>>>(fmq1_w, fq1T, HD, HD);
  k_transpose_bf16<<<tf, 256, 0, stream>>>(fmq2_w, fq2T, HD, HD);
  k_transpose_bf16<<<tf, 256, 0, stream>>>(fmk1_w, fk1T, HD, HD);
  k_transpose_bf16<<<tf, 256, 0, stream>>>(fmk2_w, fk2T, HD, HD);

  dim3 gg(HIDDIM / 128, (uint32_t)(MT / 128));
  k_gemm_bf16<<<gg, 256, 0, stream>>>(xb, wqT, nullptr, qproj, (int)MT, HIDDIM, HIDDIM);
  k_gemm_bf16<<<gg, 256, 0, stream>>>(xb, wkT, nullptr, kproj, (int)MT, HIDDIM, HIDDIM);
  k_gemm_bf16<<<gg, 256, 0, stream>>>(xb, wvT, nullptr, vproj, (int)MT, HIDDIM, HIDDIM);

  const int Mfm = (int)(MT * NHEADS);                    // 262144 head-token rows
  k_featmap<<<Mfm / 64, 256, 0, stream>>>(qproj, fq1T, fq2T, fmq1_b, fmq2_b, qfeat,
                                          0.08838834764831845f /* D^-0.5 */);
  k_featmap<<<Mfm / 64, 256, 0, stream>>>(kproj, fk1T, fk2T, fmk1_b, fmk2_b, kfeat, 1.0f);

  k_attn<<<BATCH * NHEADS, 256, 0, stream>>>(qfeat, kfeat, vproj, ofp);

  k_rmsnorm<<<Mfm / 8, 256, 0, stream>>>(ofp, gvec, onorm);

  k_gemm_bf16<<<gg, 256, 0, stream>>>(onorm, woT, out, nullptr, (int)MT, HIDDIM, HIDDIM);
}
